// Net_53609781789162
// MI455X (gfx1250) — matrix operand order repaired and checked by reading, not compiled, not run on hardware
//
#include <hip/hip_runtime.h>

// ---------------------------------------------------------------------------
// LSTM LM forward for MI455X (gfx1250, wave32).
//   Phase 1: embedding gather -> xs bf16 [4096 x 1024] (row 4095 zero pad)
//   Phase 2: x_gates = xs @ W_ih^T + (b_ih+b_hh), WMMA bf16 -> f32 [4096 x 4096]
//            (32x64 wave tile, register double-buffered k-pipeline)
//   Phase 3: persistent kernel, 128 blocks, grid barrier per step; W_hh held
//            in VGPRs (persistent-RNN), h broadcast from L2 each step
//   Phase 4: out = hs @ W_out^T + b_out, WMMA bf16 -> f32 [4095 x 1221]
// ---------------------------------------------------------------------------

typedef unsigned short u16;
typedef __attribute__((ext_vector_type(16))) __bf16 v16bf;
typedef __attribute__((ext_vector_type(8)))  float  v8f;

#define DIM_E   512
#define DIM_2E  1024
#define DIM_H   1024
#define DIM_4H  4096
#define DIM_C   1221
#define CPAD    1280      // DIM_C padded to multiple of 256
#define NBLK    128       // persistent LSTM blocks (8 hidden units each)

__device__ __forceinline__ u16 f32_to_bf16(float f) {
    unsigned int u = __float_as_uint(f);
    u += 0x7FFFu + ((u >> 16) & 1u);     // round-to-nearest-even
    return (u16)(u >> 16);
}

__device__ __forceinline__ v16bf as_v16bf(uint4 lo, uint4 hi) {
    union { struct { uint4 a, b; } s; v16bf v; } u;
    u.s.a = lo; u.s.b = hi;
    return u.v;
}

__device__ __forceinline__ float sigmoidf(float x) {
    return 1.0f / (1.0f + __expf(-x));
}

// --------------------------- init ------------------------------------------
__global__ void init_kernel(float* h_buf, u16* hs_bf, int* counter) {
    int tid = threadIdx.x;
    if (tid == 0) *counter = 0;
    for (int i = tid; i < 2 * DIM_H; i += 256) h_buf[i] = 0.0f;
    // zero hs pad row (m = 4095) so the output GEMM reads clean bf16
    for (int i = tid; i < DIM_H; i += 256) hs_bf[(size_t)4095 * DIM_H + i] = 0;
}

// --------------------------- embedding -------------------------------------
__global__ void embed_kernel(const int* __restrict__ x,
                             const float* __restrict__ emb,
                             u16* __restrict__ xs, int N, int T) {
    int row = blockIdx.x;                       // 0..4095
    int tid = threadIdx.x;
    u16* dst = xs + (size_t)row * DIM_2E;
    if (row >= N) {
        for (int c = tid; c < DIM_2E; c += 256) dst[c] = 0;
        return;
    }
    const float* e0 = emb + (size_t)x[row]   * DIM_E;
    const float* e1 = emb + (size_t)x[T - 1] * DIM_E;
    for (int c = tid; c < DIM_2E; c += 256) {
        float v = (c < DIM_E) ? e0[c] : e1[c - DIM_E];
        dst[c] = f32_to_bf16(v);
    }
}

// --------------------------- f32 -> bf16 convert (with row padding) --------
__global__ void cvt_pad_kernel(const float* __restrict__ src,
                               u16* __restrict__ dst,
                               int rows_valid, int cols, long total) {
    long i = (long)blockIdx.x * blockDim.x + threadIdx.x;
    if (i >= total) return;
    long r = i / cols;
    long c = i - r * cols;
    dst[i] = (r < rows_valid) ? f32_to_bf16(src[r * cols + c]) : (u16)0;
}

__global__ void bias_kernel(const float* __restrict__ a,
                            const float* __restrict__ b,
                            float* __restrict__ o, int n) {
    int i = blockIdx.x * blockDim.x + threadIdx.x;
    if (i < n) o[i] = a[i] + b[i];
}

// --------------------------- bf16 WMMA GEMM --------------------------------
// C[m,n] = sum_k A[m,k] * B[n,k] + bias[n]     (A, B row-major bf16, C f32)
// Block: 256 threads = 8 waves, tile 64(M) x 256(N); wave tile 32 x 64.
// Register double-buffered along K so loads of tile k+1 overlap the 8 WMMAs
// of tile k (single s_wait_loadcnt per k-step, after the WMMAs).

struct Frag {
    uint4 a[2][2];   // two 16x32 A subtiles
    uint4 b[4][2];   // four 32x16 B subtiles
};

__device__ __forceinline__ void load_frag(Frag& f,
                                          const u16* ap0, const u16* ap1,
                                          const u16* const bp[4], int k0) {
    f.a[0][0] = *(const uint4*)(ap0 + k0);
    f.a[0][1] = *(const uint4*)(ap0 + k0 + 16);
    f.a[1][0] = *(const uint4*)(ap1 + k0);
    f.a[1][1] = *(const uint4*)(ap1 + k0 + 16);
#pragma unroll
    for (int s = 0; s < 4; ++s) {
        f.b[s][0] = *(const uint4*)(bp[s] + k0);
        f.b[s][1] = *(const uint4*)(bp[s] + k0 + 16);
    }
}

__device__ __forceinline__ void mma_frag(v8f acc[2][4], const Frag& f) {
#pragma unroll
    for (int i = 0; i < 2; ++i) {
        v16bf av = as_v16bf(f.a[i][0], f.a[i][1]);
#pragma unroll
        for (int s = 0; s < 4; ++s) {
            v16bf bv = as_v16bf(f.b[s][0], f.b[s][1]);
            acc[i][s] = __builtin_amdgcn_wmma_f32_16x16x32_bf16(
                false, av, false, bv, (short)0, acc[i][s], false, false);
        }
    }
}

__global__ __launch_bounds__(256)
void gemm_bf16_wmma(const u16* __restrict__ A, const u16* __restrict__ B,
                    const float* __restrict__ bias, float* __restrict__ C,
                    int K, int ldc, int m_valid, int n_valid) {
    const int lane = threadIdx.x & 31;
    const int w    = threadIdx.x >> 5;          // 0..7
    const int m0   = blockIdx.y * 64 + (w & 1) * 32;
    const int n0   = blockIdx.x * 256 + (w >> 1) * 64;
    const int kA   = (lane >> 4) * 8;           // A: K halves [kA,kA+8) & [kA+16,kA+24)
    const int kB   = (lane >> 4) * 8;           // B: K halves [kB,kB+8) & [kB+16,kB+24)

    const u16* ap0 = A + (size_t)(m0 + (lane & 15)) * K + kA;
    const u16* ap1 = ap0 + (size_t)16 * K;
    const u16* bp[4];
#pragma unroll
    for (int s = 0; s < 4; ++s)
        bp[s] = B + (size_t)(n0 + s * 16 + (lane & 15)) * K + kB;

    v8f acc[2][4] = {{v8f{}, v8f{}, v8f{}, v8f{}}, {v8f{}, v8f{}, v8f{}, v8f{}}};

    Frag cur, nxt;
    load_frag(cur, ap0, ap1, bp, 0);
    int k0 = 0;
    for (; k0 + 32 < K; k0 += 32) {
        load_frag(nxt, ap0, ap1, bp, k0 + 32);   // issue next tile's loads
        mma_frag(acc, cur);                      // 8 WMMAs on current tile
        cur = nxt;
    }
    mma_frag(acc, cur);

    const int rbase = (lane >> 4) * 8;           // C: vgpr r -> M = r + rbase
    const int ncol  = lane & 15;
#pragma unroll
    for (int i = 0; i < 2; ++i) {
#pragma unroll
        for (int s = 0; s < 4; ++s) {
            int n = n0 + s * 16 + ncol;
            if (n >= n_valid) continue;
            float bv = bias[n];
#pragma unroll
            for (int r = 0; r < 8; ++r) {
                int m = m0 + i * 16 + rbase + r;
                if (m < m_valid) C[(size_t)m * ldc + n] = acc[i][s][r] + bv;
            }
        }
    }
}

// --------------------------- persistent LSTM -------------------------------
// 128 blocks x 256 threads. Block b owns hidden units [b*8, b*8+8) and their
// 4 gate rows (i,f,g,o at j, j+1024, j+2048, j+3072) = 32 rows of W_hh.
// Each thread keeps its 128 weights in VGPRs (32 float4) for the whole scan:
// zero weight traffic per step; only the 4 KB h broadcast comes from L2.
// 8 lanes cooperate per row, interleaved at float4 granularity so each wave
// load instruction covers contiguous 128B segments.
__global__ __launch_bounds__(256)
void lstm_kernel(const float* __restrict__ x_gates,
                 const float* __restrict__ W_hh,
                 float* __restrict__ h_buf,          // [2][1024]
                 u16*   __restrict__ hs_bf,          // [4096][1024]
                 int*   __restrict__ counter, int nsteps) {
    __shared__ float sgate[32];
    __shared__ float c_s[8];

    const int tid  = threadIdx.x;
    const int b    = blockIdx.x;                // 0..127
    const int j0   = b * 8;
    const int rloc = tid >> 3;                  // 0..31 local gate row
    const int part = tid & 7;                   // 8 lanes per row dot product
    const int gate = rloc >> 3;                 // 0:i 1:f 2:g 3:o
    const int unit = rloc & 7;
    const int grow = gate * DIM_H + j0 + unit;  // global gate row in [0,4096)

    // preload this thread's 128 weights: k = i*32 + part*4 .. +4
    const float4* wr = (const float4*)(W_hh + (size_t)grow * DIM_H) + part;
    float4 wreg[32];
#pragma unroll
    for (int i = 0; i < 32; ++i) wreg[i] = wr[i * 8];

    if (tid < 8) c_s[tid] = 0.0f;
    __syncthreads();

    for (int t = 0; t < nsteps; ++t) {
        const float4* h4 = (const float4*)(h_buf + (t & 1) * DIM_H) + part;
        float acc = 0.0f;
#pragma unroll
        for (int i = 0; i < 32; ++i) {          // 128 MACs from register weights
            float4 hv = h4[i * 8];
            acc = fmaf(wreg[i].x, hv.x, acc);
            acc = fmaf(wreg[i].y, hv.y, acc);
            acc = fmaf(wreg[i].z, hv.z, acc);
            acc = fmaf(wreg[i].w, hv.w, acc);
        }
        acc += __shfl_xor(acc, 1, 32);          // reduce over 8 part-lanes
        acc += __shfl_xor(acc, 2, 32);
        acc += __shfl_xor(acc, 4, 32);
        if (part == 0)
            sgate[rloc] = acc + x_gates[(size_t)t * DIM_4H + grow];
        __syncthreads();

        if (tid < 8) {
            float gi = sigmoidf(sgate[tid]);
            float gf = sigmoidf(sgate[8 + tid]);
            float gg = tanhf(sgate[16 + tid]);
            float go = sigmoidf(sgate[24 + tid]);
            float c  = gf * c_s[tid] + gi * gg;
            c_s[tid] = c;
            float h  = go * tanhf(c);
            h_buf[((t + 1) & 1) * DIM_H + j0 + tid] = h;
            hs_bf[(size_t)t * DIM_H + j0 + tid] = f32_to_bf16(h);
        }
        __syncthreads();

        // device-scope grid barrier between steps
        if (tid == 0) {
            __threadfence();
            __hip_atomic_fetch_add(counter, 1, __ATOMIC_RELEASE,
                                   __HIP_MEMORY_SCOPE_AGENT);
            const int target = NBLK * (t + 1);
            while (__hip_atomic_load(counter, __ATOMIC_ACQUIRE,
                                     __HIP_MEMORY_SCOPE_AGENT) < target)
                __builtin_amdgcn_s_sleep(2);
            __threadfence();
        }
        __syncthreads();
    }
}

// --------------------------- launch ----------------------------------------
extern "C" void kernel_launch(void* const* d_in, const int* in_sizes, int n_in,
                              void* d_out, int out_size, void* d_ws, size_t ws_size,
                              hipStream_t stream) {
    const int*   x      = (const int*)  d_in[0];
    const float* embT   = (const float*)d_in[1];
    const float* W_ih   = (const float*)d_in[2];
    const float* W_hh   = (const float*)d_in[3];
    const float* b_ih   = (const float*)d_in[4];
    const float* b_hh   = (const float*)d_in[5];
    const float* W_out  = (const float*)d_in[6];
    const float* b_out  = (const float*)d_in[7];
    float*       out    = (float*)d_out;

    const int T = in_sizes[0];                  // 4096
    const int N = T - 1;                        // 4095 steps
    const int Mpad = 4096;

    // workspace layout (~91 MB)
    char* ws = (char*)d_ws;
    u16* xs_bf    = (u16*)ws;            ws += (size_t)Mpad * DIM_2E * 2;   // 8 MB
    u16* wih_bf   = (u16*)ws;            ws += (size_t)DIM_4H * DIM_2E * 2; // 8 MB
    u16* wout_bf  = (u16*)ws;            ws += (size_t)CPAD * DIM_H * 2;    // 2.5 MB
    u16* hs_bf    = (u16*)ws;            ws += (size_t)Mpad * DIM_H * 2;    // 8 MB
    float* gbias  = (float*)ws;          ws += (size_t)DIM_4H * 4;
    float* xgates = (float*)ws;          ws += (size_t)Mpad * DIM_4H * 4;   // 64 MB
    float* h_buf  = (float*)ws;          ws += 2 * DIM_H * 4;
    int* counter  = (int*)ws;

    // Phase 0: state init (re-run every graph replay)
    init_kernel<<<1, 256, 0, stream>>>(h_buf, hs_bf, counter);

    // Phase 1: embedding gather + operand conversion
    embed_kernel<<<Mpad, 256, 0, stream>>>(x, embT, xs_bf, N, T);
    {
        long tot = (long)DIM_4H * DIM_2E;
        cvt_pad_kernel<<<(tot + 255) / 256, 256, 0, stream>>>(W_ih, wih_bf,
                                                              DIM_4H, DIM_2E, tot);
    }
    {
        long tot = (long)CPAD * DIM_H;
        cvt_pad_kernel<<<(tot + 255) / 256, 256, 0, stream>>>(W_out, wout_bf,
                                                              DIM_C, DIM_H, tot);
    }
    bias_kernel<<<(DIM_4H + 255) / 256, 256, 0, stream>>>(b_ih, b_hh, gbias, DIM_4H);

    // Phase 2: x_gates = xs @ W_ih^T + (b_ih + b_hh)   [4096 x 4096]
    {
        dim3 grid(DIM_4H / 256, Mpad / 64);
        gemm_bf16_wmma<<<grid, 256, 0, stream>>>(xs_bf, wih_bf, gbias, xgates,
                                                 DIM_2E, DIM_4H, Mpad, DIM_4H);
    }

    // Phase 3: recurrent scan (persistent, grid-synced per step)
    lstm_kernel<<<NBLK, 256, 0, stream>>>(xgates, W_hh, h_buf, hs_bf, counter, N);

    // Phase 4: out = hs @ W_out^T + b_out   [4095 x 1221]
    {
        dim3 grid(CPAD / 256, Mpad / 64);
        gemm_bf16_wmma<<<grid, 256, 0, stream>>>(hs_bf, wout_bf, b_out, out,
                                                 DIM_H, DIM_C, N, DIM_C);
    }
}
